// RNN_72490458022185
// MI455X (gfx1250) — matrix operand order repaired and checked by reading, not compiled, not run on hardware
//
#include <hip/hip_runtime.h>
#include <stdint.h>

#define B_ 128
#define T_ 2048
#define I_ 128
#define H_ 512
#define C_ 128

typedef __bf16 bf16;
typedef __attribute__((ext_vector_type(4)))  bf16  v4bf;
typedef __attribute__((ext_vector_type(8)))  bf16  v8bf;
typedef __attribute__((ext_vector_type(16))) bf16  v16bf;
typedef __attribute__((ext_vector_type(8)))  float v8f;

__device__ __forceinline__ v16bf join8(v8bf lo, v8bf hi) {
  return __builtin_shufflevector(lo, hi, 0, 1, 2, 3, 4, 5, 6, 7,
                                         8, 9, 10, 11, 12, 13, 14, 15);
}

// A fragment (16x32 bf16, ISA 7.12.2): lane holds row (lane&15);
// K elements [(lane>>4)*8 .. +7] and [16+(lane>>4)*8 .. +7]  -> 2x16B loads.
__device__ __forceinline__ v16bf fragA_bf16(const bf16* base, int stride,
                                            int k0, int lane) {
  const int m = lane & 15, half = lane >> 4;
  const bf16* p0 = base + m * stride + k0 + half * 8;
  return join8(*(const v8bf*)p0, *(const v8bf*)(p0 + 16));
}

// B fragment (32x16 bf16). out = h @ W^T  =>  B[k][n] = W[n][k], so each lane
// reads K [(lane>>4)*8 .. +7] and [16+(lane>>4)*8 .. +7] from row (n0 + lane&15) of row-major W.
// Non-temporal: the per-step weight stream (104KB/wave) can never live in the
// 64KB WGP$, so bypass near cache; stays resident in the 192MB L2.
__device__ __forceinline__ v16bf fragB_nt(const bf16* __restrict__ W,
                                          int stride, int n0, int k0,
                                          int lane) {
  const int n = lane & 15, half = lane >> 4;
  const bf16* p = W + (size_t)(n0 + n) * stride + k0 + half * 8;
  v8bf lo = __builtin_nontemporal_load((const v8bf*)p);
  v8bf hi = __builtin_nontemporal_load((const v8bf*)(p + 16));
  return join8(lo, hi);
}

__device__ __forceinline__ v8f wmma_bf16(v16bf a, v16bf b, v8f c) {
  // (neg_a, A, neg_b, B, c_mod, C, reuse_a, reuse_b)
  return __builtin_amdgcn_wmma_f32_16x16x32_bf16(false, a, false, b,
                                                 (short)0, c, false, false);
}

// CDNA5 async global->LDS copy (16B per lane) + wait, per CDNA5 ISA.
__device__ __forceinline__ void async_copy16(uint32_t lds_addr,
                                             const void* gaddr) {
  asm volatile("global_load_async_to_lds_b128 %0, %1, off"
               :: "v"(lds_addr), "v"(gaddr)
               : "memory");
}
__device__ __forceinline__ void wait_async0() {
  asm volatile("s_wait_asynccnt 0" ::: "memory");
}

__global__ __launch_bounds__(512, 1)
void rnn_persistent(const float* __restrict__ x,
                    const float* __restrict__ b0,
                    const float* __restrict__ b1,
                    const float* __restrict__ fcb,
                    const bf16* __restrict__ Wih0,
                    const bf16* __restrict__ Whh0,
                    const bf16* __restrict__ Wih1,
                    const bf16* __restrict__ Whh1,
                    const bf16* __restrict__ Wfc,
                    float* __restrict__ out) {
  __shared__ float xbuf[16][I_];          // f32 x_t staging (async-filled)
  __shared__ bf16  xb[16][I_];            // bf16 x_t (fragment-ready)
  __shared__ bf16  h0buf[2][16][H_];      // double-buffered recurrent state
  __shared__ bf16  h1buf[2][16][H_];

  const int tid  = threadIdx.x;           // 0..511 (16 wave32 waves)
  const int wave = tid >> 5;
  const int lane = tid & 31;
  const int half = lane >> 4;
  const int nn   = lane & 15;
  const int b_base = blockIdx.x * 16;     // this WG's batch rows

  // zero initial hidden state (h0buf[0], h1buf[0])
  for (int i = tid; i < 16 * H_; i += 512) {
    (&h0buf[0][0][0])[i] = (bf16)0.0f;
    (&h1buf[0][0][0])[i] = (bf16)0.0f;
  }

  // each wave owns two adjacent 16-wide N-tiles -> A fragments reused 2x
  const int n0 = wave * 32;
  const float bias0a = b0[n0 + nn],      bias1a = b1[n0 + nn];
  const float bias0b = b0[n0 + 16 + nn], bias1b = b1[n0 + 16 + nn];

  // async prefetch x_{t=0}: 16 rows * 128 f32 = 512 lanes * 16B
  {
    const int m = tid >> 5, ch = tid & 31;
    const uint32_t lds = (uint32_t)(uintptr_t)&xbuf[m][ch * 4];
    const float* g = x + ((size_t)(b_base + m) * T_ + 0) * I_ + ch * 4;
    async_copy16(lds, g);
  }

  int cur = 0;
#pragma unroll 1
  for (int t = 0; t < T_; ++t) {
    const int nxt = cur ^ 1;

    // each thread converts exactly the 16B its own async op DMA'd
    // (flat index tid*4 == m*128 + ch*4), so its own asynccnt wait suffices;
    // the top barrier then publishes the bf16 tile to all waves.
    wait_async0();
    {
      const float* pf = &xbuf[0][0] + tid * 4;
      v4bf o;
      o[0] = (bf16)pf[0]; o[1] = (bf16)pf[1];
      o[2] = (bf16)pf[2]; o[3] = (bf16)pf[3];
      *(v4bf*)(&xb[0][0] + tid * 4) = o;
    }
    __syncthreads();                      // xb ready; states settled

    // ---- layer 0: h0' = tanh(x_t @ Wih0^T + h0 @ Whh0^T + b0) ----
    v8f a0 = {}, a1 = {};
#pragma unroll
    for (int kb = 0; kb < I_ / 32; ++kb) {
      const int k0 = kb * 32;
      v16bf A  = fragA_bf16(&xb[0][0], I_, k0, lane);
      v16bf B0 = fragB_nt(Wih0, I_, n0, k0, lane);
      v16bf B1 = fragB_nt(Wih0, I_, n0 + 16, k0, lane);
      a0 = wmma_bf16(A, B0, a0);
      a1 = wmma_bf16(A, B1, a1);
    }
#pragma unroll
    for (int kb = 0; kb < H_ / 32; ++kb) {
      const int k0 = kb * 32;
      v16bf A  = fragA_bf16(&h0buf[cur][0][0], H_, k0, lane);
      v16bf B0 = fragB_nt(Whh0, H_, n0, k0, lane);
      v16bf B1 = fragB_nt(Whh0, H_, n0 + 16, k0, lane);
      a0 = wmma_bf16(A, B0, a0);
      a1 = wmma_bf16(A, B1, a1);
    }
#pragma unroll
    for (int j = 0; j < 8; ++j) {
      h0buf[nxt][half * 8 + j][n0 + nn]      = (bf16)tanhf(a0[j] + bias0a);
      h0buf[nxt][half * 8 + j][n0 + 16 + nn] = (bf16)tanhf(a1[j] + bias0b);
    }
    __syncthreads();                      // h0[nxt] complete; xbuf free

    // prefetch x_{t+1} overlapped with layer-1 GEMMs
    if (t + 1 < T_) {
      const int m = tid >> 5, ch = tid & 31;
      const uint32_t lds = (uint32_t)(uintptr_t)&xbuf[m][ch * 4];
      const float* g = x + ((size_t)(b_base + m) * T_ + (t + 1)) * I_ + ch * 4;
      async_copy16(lds, g);
    }

    // ---- layer 1: h1' = tanh(h0' @ Wih1^T + h1 @ Whh1^T + b1) ----
    v8f c0 = {}, c1 = {};
#pragma unroll
    for (int kb = 0; kb < H_ / 32; ++kb) {
      const int k0 = kb * 32;
      v16bf A  = fragA_bf16(&h0buf[nxt][0][0], H_, k0, lane);
      v16bf B0 = fragB_nt(Wih1, H_, n0, k0, lane);
      v16bf B1 = fragB_nt(Wih1, H_, n0 + 16, k0, lane);
      c0 = wmma_bf16(A, B0, c0);
      c1 = wmma_bf16(A, B1, c1);
    }
#pragma unroll
    for (int kb = 0; kb < H_ / 32; ++kb) {
      const int k0 = kb * 32;
      v16bf A  = fragA_bf16(&h1buf[cur][0][0], H_, k0, lane);
      v16bf B0 = fragB_nt(Whh1, H_, n0, k0, lane);
      v16bf B1 = fragB_nt(Whh1, H_, n0 + 16, k0, lane);
      c0 = wmma_bf16(A, B0, c0);
      c1 = wmma_bf16(A, B1, c1);
    }
#pragma unroll
    for (int j = 0; j < 8; ++j) {
      h1buf[nxt][half * 8 + j][n0 + nn]      = (bf16)tanhf(c0[j] + bias1a);
      h1buf[nxt][half * 8 + j][n0 + 16 + nn] = (bf16)tanhf(c1[j] + bias1b);
    }
    cur = nxt;                            // loop-top barrier orders next reads
  }

  __syncthreads();                        // final h1 visible

  // ---- FC: out = h1 @ fc_w^T + fc_b   (C=128 -> waves 0..7, one tile each)
  if (wave < 8) {
    const int col0 = wave * 16;
    v8f accf = {};
#pragma unroll
    for (int kb = 0; kb < H_ / 32; ++kb) {
      v16bf A = fragA_bf16(&h1buf[cur][0][0], H_, kb * 32, lane);
      v16bf B = fragB_nt(Wfc, H_, col0, kb * 32, lane);
      accf = wmma_bf16(A, B, accf);
    }
    const float bc = fcb[col0 + nn];
#pragma unroll
    for (int j = 0; j < 8; ++j) {
      out[(size_t)(b_base + half * 8 + j) * C_ + col0 + nn] = accf[j] + bc;
    }
  }
}

__global__ void cvt_f32_to_bf16(const float* __restrict__ s,
                                bf16* __restrict__ d, int n) {
  const int i = blockIdx.x * blockDim.x + threadIdx.x;
  if (i < n) d[i] = (bf16)s[i];
}

extern "C" void kernel_launch(void* const* d_in, const int* in_sizes, int n_in,
                              void* d_out, int out_size, void* d_ws,
                              size_t ws_size, hipStream_t stream) {
  const float* x     = (const float*)d_in[0];
  const float* Wih0f = (const float*)d_in[1];
  const float* Whh0f = (const float*)d_in[2];
  const float* b0    = (const float*)d_in[3];
  const float* Wih1f = (const float*)d_in[4];
  const float* Whh1f = (const float*)d_in[5];
  const float* b1    = (const float*)d_in[6];
  const float* fcwf  = (const float*)d_in[7];
  const float* fcb   = (const float*)d_in[8];

  // bf16 weight cache in workspace (~1.75 MB total, L2-resident)
  bf16* ws   = (bf16*)d_ws;
  bf16* Wih0 = ws;
  bf16* Whh0 = Wih0 + (size_t)H_ * I_;
  bf16* Wih1 = Whh0 + (size_t)H_ * H_;
  bf16* Whh1 = Wih1 + (size_t)H_ * H_;
  bf16* Wfc  = Whh1 + (size_t)H_ * H_;

  cvt_f32_to_bf16<<<(H_ * I_ + 255) / 256, 256, 0, stream>>>(Wih0f, Wih0, H_ * I_);
  cvt_f32_to_bf16<<<(H_ * H_ + 255) / 256, 256, 0, stream>>>(Whh0f, Whh0, H_ * H_);
  cvt_f32_to_bf16<<<(H_ * H_ + 255) / 256, 256, 0, stream>>>(Wih1f, Wih1, H_ * H_);
  cvt_f32_to_bf16<<<(H_ * H_ + 255) / 256, 256, 0, stream>>>(Whh1f, Whh1, H_ * H_);
  cvt_f32_to_bf16<<<(C_ * H_ + 255) / 256, 256, 0, stream>>>(fcwf, Wfc, C_ * H_);

  rnn_persistent<<<B_ / 16, 512, 0, stream>>>(x, b0, b1, fcb, Wih0, Whh0,
                                              Wih1, Whh1, Wfc, (float*)d_out);
}
